// BehaviorEngine_26259430047890
// MI455X (gfx1250) — matrix operand order repaired and checked by reading, not compiled, not run on hardware
//
#include <hip/hip_runtime.h>

// CDNA5 (gfx1250) wave32 WMMA MLP, weights-as-A / activations-as-B formulation:
//   h^T[64,N16] = W1^T[64,32] @ x^T[32,N16]   (pot column folded into C-init)
//   o^T[32,N16] = W2^T[32,64] @ relu(h)^T     (half-wave swap, no LDS transpose)
//   out = o + 0.3*s   (single LDS transpose at the end for coalesced stores)

typedef __attribute__((ext_vector_type(16))) _Float16 v16h;
typedef __attribute__((ext_vector_type(8)))  float    v8f;
typedef __attribute__((ext_vector_type(4)))  float    vf4;
typedef __attribute__((ext_vector_type(2)))  __fp16   fp16x2;

#define EPSV   1e-8f
#define ALPHA  0.3f
#define PATW   0.1f

#define WAVES_PER_BLOCK 8
#define BLOCK_THREADS   (WAVES_PER_BLOCK * 32)
#define DDIM  32
#define HID   64
#define KP    36   // padded f32 row stride: 16B-aligned rows, conflict-free banks

__global__ __launch_bounds__(BLOCK_THREADS)
void behavior_mlp_wmma(const float* __restrict__ grid,   // [N,32]
                       const float* __restrict__ pot,    // [N]
                       const float* __restrict__ W1,     // [33,64]
                       const float* __restrict__ b1,     // [64]
                       const float* __restrict__ W2,     // [64,32]
                       const float* __restrict__ b2,     // [32]
                       float* __restrict__ out,          // [N,32]
                       int nTiles)                       // N/16
{
    __shared__ __attribute__((aligned(16))) float obuf[WAVES_PER_BLOCK][16 * KP];

    const int lane  = threadIdx.x & 31;
    const int w     = threadIdx.x >> 5;
    const int bcol  = lane & 15;   // batch row within tile == B/C/D column (N)
    const int khalf = lane >> 4;   // K/M sub-range selector per WMMA layouts

    // ---------- Persistent A-fragments: transposed weights -------------------
    // 16-bit A 16x32 frags in this kernel's contiguous K order (same as B), own-lane K set: {khalf*16+0..15}
    // Layer 1: A = W1^T (hidden x input), 4 M-tiles, single K=32 step.
    v16h A1[4];
    #pragma unroll
    for (int mt = 0; mt < 4; ++mt) {
        _Float16* f = (_Float16*)&A1[mt];
        const int m = mt * 16 + bcol;                 // hidden index (A row)
        #pragma unroll
        for (int v = 0; v < 8; ++v) {
            const int k = khalf * 16 + 2 * v;
            f[2 * v]     = (_Float16)W1[k * HID + m];
            f[2 * v + 1] = (_Float16)W1[(k + 1) * HID + m];
        }
    }
    // Layer 2: A = W2^T (out x hidden), 2 M-tiles x 2 K-steps.
    v16h A2[2][2];
    #pragma unroll
    for (int ot = 0; ot < 2; ++ot) {
        #pragma unroll
        for (int ks = 0; ks < 2; ++ks) {
            _Float16* f = (_Float16*)&A2[ot][ks];
            const int m = ot * 16 + bcol;             // out-dim index (A row)
            #pragma unroll
            for (int v = 0; v < 8; ++v) {
                const int k = ks * 32 + khalf * 16 + 2 * v;
                f[2 * v]     = (_Float16)W2[k * DDIM + m];
                f[2 * v + 1] = (_Float16)W2[(k + 1) * DDIM + m];
            }
        }
    }
    // Per-lane bias vectors along the C/D M axis (m = base + 8*khalf + g).
    float b1v[4][8], w1p[4][8], b2v[2][8];
    #pragma unroll
    for (int mt = 0; mt < 4; ++mt) {
        *(vf4*)&b1v[mt][0] = *(const vf4*)(b1 + mt * 16 + 8 * khalf);
        *(vf4*)&b1v[mt][4] = *(const vf4*)(b1 + mt * 16 + 8 * khalf + 4);
        *(vf4*)&w1p[mt][0] = *(const vf4*)(W1 + 32 * HID + mt * 16 + 8 * khalf);
        *(vf4*)&w1p[mt][4] = *(const vf4*)(W1 + 32 * HID + mt * 16 + 8 * khalf + 4);
    }
    #pragma unroll
    for (int ot = 0; ot < 2; ++ot) {
        *(vf4*)&b2v[ot][0] = *(const vf4*)(b2 + ot * 16 + 8 * khalf);
        *(vf4*)&b2v[ot][4] = *(const vf4*)(b2 + ot * 16 + 8 * khalf + 4);
    }

    const int waveGlobal = blockIdx.x * WAVES_PER_BLOCK + w;
    const int waveStride = gridDim.x * WAVES_PER_BLOCK;

    for (int tile = waveGlobal; tile < nTiles; tile += waveStride) {
        const int row = tile * 16 + bcol;             // this lane's batch row
        const float* gp = grid + (size_t)row * DDIM + khalf * 16;

        // ---- x = s + 0.1 * s^3/(||s^3||+eps) * ||s||, packed as B-frag ------
        float s[16], c[16];
        #pragma unroll
        for (int q = 0; q < 4; ++q)
            *(vf4*)&s[4 * q] = *(const vf4*)(gp + 4 * q);
        float ssum = 0.0f, csum = 0.0f;
        #pragma unroll
        for (int j = 0; j < 16; ++j) {
            c[j] = s[j] * s[j] * s[j];
            ssum += s[j] * s[j];
            csum += c[j] * c[j];
        }
        ssum += __shfl_xor(ssum, 16, 32);             // combine the two d-halves
        csum += __shfl_xor(csum, 16, 32);
        const float scale = PATW * __fsqrt_rn(ssum) / (__fsqrt_rn(csum) + EPSV);
        v16h xb;                                      // B-frag: VGPR v = K{2v,2v+1}
        int* xp = (int*)&xb;
        #pragma unroll
        for (int v = 0; v < 8; ++v)
            xp[v] = __builtin_bit_cast(int, __builtin_amdgcn_cvt_pkrtz(
                        s[2 * v]     + scale * c[2 * v],
                        s[2 * v + 1] + scale * c[2 * v + 1]));
        const float potv = pot[row];

        // ---- Layer 1: h^T tiles; pot column via rank-1 C-init ---------------
        v8f hacc[4];
        #pragma unroll
        for (int mt = 0; mt < 4; ++mt) {
            v8f cc;
            #pragma unroll
            for (int g = 0; g < 8; ++g)
                cc[g] = __builtin_fmaf(potv, w1p[mt][g], b1v[mt][g]);
            hacc[mt] = __builtin_amdgcn_wmma_f32_16x16x32_f16(
                false, A1[mt], false, xb, (short)0, cc, false, false);
        }

        // ---- pack to f16, single-instruction packed ReLU --------------------
        // relu(cvt(x)) == cvt(relu(x)): conversion is monotone, sign-preserving.
        int ownp[4][4];
        #pragma unroll
        for (int mt = 0; mt < 4; ++mt) {
            #pragma unroll
            for (int v = 0; v < 4; ++v) {
                const int pi = __builtin_bit_cast(
                    int, __builtin_amdgcn_cvt_pkrtz(hacc[mt][2 * v],
                                                    hacc[mt][2 * v + 1]));
                int po;
                asm("v_pk_max_num_f16 %0, %1, 0" : "=v"(po) : "v"(pi));
                ownp[mt][v] = po;
            }
        }

        // ---- half-wave swap: send only what the partner consumes ------------
        // B-frag(ks): K = ks*32 + khalf*16 + {0..15}. khalf=0 lanes take
        // mt=2ks (own m=0..7, partner m=8..15); khalf=1 lanes take mt=2ks+1
        // (partner m=16..23, own m=24..31). 8 bpermutes total.
        v16h hbf[2];
        #pragma unroll
        for (int ks = 0; ks < 2; ++ks) {
            int* hp = (int*)&hbf[ks];
            #pragma unroll
            for (int v = 0; v < 4; ++v) {
                const int snd = khalf ? ownp[2 * ks][v] : ownp[2 * ks + 1][v];
                const int rcv = __shfl_xor(snd, 16, 32);
                hp[v]     = khalf ? rcv : ownp[2 * ks][v];
                hp[v + 4] = khalf ? ownp[2 * ks + 1][v] : rcv;
            }
        }

        // ---- Layer 2: o^T = W2^T @ relu(h)^T + b2 ---------------------------
        v8f oacc[2];
        #pragma unroll
        for (int ot = 0; ot < 2; ++ot) {
            v8f cc;
            #pragma unroll
            for (int g = 0; g < 8; ++g) cc[g] = b2v[ot][g];
            cc = __builtin_amdgcn_wmma_f32_16x16x32_f16(
                false, A2[ot][0], false, hbf[0], (short)0, cc, false, false);
            cc = __builtin_amdgcn_wmma_f32_16x16x32_f16(
                false, A2[ot][1], false, hbf[1], (short)0, cc, false, false);
            oacc[ot] = cc;
        }

        // ---- Epilogue: transpose D2 via LDS, add 0.3*s, coalesced stores ----
        float* ob = obuf[w];
        #pragma unroll
        for (int ot = 0; ot < 2; ++ot)
            #pragma unroll
            for (int g = 0; g < 8; ++g)
                ob[bcol * KP + ot * 16 + 8 * khalf + g] = oacc[ot][g];
        asm volatile("s_wait_dscnt 0" ::: "memory");

        float r[16];
        const float* orow = &obuf[w][bcol * KP + khalf * 16];
        #pragma unroll
        for (int q = 0; q < 4; ++q)
            *(vf4*)&r[4 * q] = *(const vf4*)(orow + 4 * q);
        #pragma unroll
        for (int j = 0; j < 16; ++j)
            r[j] = __builtin_fmaf(ALPHA, s[j], r[j]);
        float* op = out + (size_t)row * DDIM + khalf * 16;
        #pragma unroll
        for (int q = 0; q < 4; ++q)
            *(vf4*)(op + 4 * q) = *(const vf4*)&r[4 * q];
    }
}

extern "C" void kernel_launch(void* const* d_in, const int* in_sizes, int n_in,
                              void* d_out, int out_size, void* d_ws, size_t ws_size,
                              hipStream_t stream) {
    const float* grid = (const float*)d_in[0];   // [H,W,32]
    const float* pot  = (const float*)d_in[1];   // [H,W]
    const float* W1   = (const float*)d_in[2];   // [33,64]
    const float* b1   = (const float*)d_in[3];   // [64]
    const float* W2   = (const float*)d_in[4];   // [64,32]
    const float* b2   = (const float*)d_in[5];   // [32]
    float* out        = (float*)d_out;

    const int rows   = in_sizes[0] / DDIM;       // 1,048,576
    const int nTiles = rows / 16;                // 65,536
    int blocks = (nTiles + WAVES_PER_BLOCK - 1) / WAVES_PER_BLOCK;
    if (blocks > 4096) blocks = 4096;            // grid-stride covers the rest

    behavior_mlp_wmma<<<blocks, BLOCK_THREADS, 0, stream>>>(
        grid, pot, W1, b1, W2, b2, out, nTiles);
}
